// GAST_7017976561732
// MI455X (gfx1250) — matrix operand order repaired and checked by reading, not compiled, not run on hardware
//
#include <hip/hip_runtime.h>
#include <math.h>

// ---------------- problem constants (match reference) ----------------
#define BATCH 256
#define CIN   200
#define CPAD  224      // CIN padded to multiple of 32
#define NNODE 121
#define DDIM  64
#define GHEAD 8
#define NLAY  4
#define THEAD 8
#define FFDIM 128
#define NCLS  16
#define MROWS (BATCH*NNODE)   // 30976, multiple of 32

typedef __attribute__((ext_vector_type(16))) _Float16 v16h;
typedef __attribute__((ext_vector_type(8)))  _Float16 v8h;
typedef __attribute__((ext_vector_type(8)))  float    v8f;

// ---------------------------------------------------------------------
// WMMA GEMM, f16 operands, f32 accumulate:
//   OUT[m,n] = act( sum_k A[m,k] * W[n,k] + bias[n] )
// A: row-major [M, lda] f16; W: row-major [Nc, ldw] f16 (B = W^T).
// Caller guarantees: K%32==0, lda/ldw%8==0, M%(16*MT)==0, Nc%(16*NT)==0.
// One wave per (16*MT)x(16*NT) tile; pure b128 loads + v_wmma in the loop.
// Fragment layouts per CDNA5 ISA 7.12.2 (A: halves 0..7->K=kb..kb+7,
// 8..15->K=kb+16..kb+23, kb=(L>=16)?8:0; B: halves 0..7->K=kb2..kb2+7, 8..15->K=kb2+16..kb2+23,
// kb2=(L>=16)?8:0; C/D: elem j -> row j+((L>=16)?8:0), col L&15).
// ACT: 0=none 1=gelu(erf) 2=relu 3=sigmoid.  OMODE bit0: f32 out, bit1: f16 out.
// ---------------------------------------------------------------------
template <int NT, int MT, int ACT, int OMODE>
__global__ __launch_bounds__(32, 1)
void gast_gemm_wmma(const _Float16* __restrict__ A, const _Float16* __restrict__ W,
                    const float* __restrict__ bias, float* __restrict__ OUT32,
                    _Float16* __restrict__ OUT16, int K, int lda, int ldw, int ldo)
{
    const int lane = threadIdx.x;
    const int tm = blockIdx.y * (16 * MT);
    const int tn = blockIdx.x * (16 * NT);
    const int kbaseA = (lane >> 4) * 8;
    const int kbaseB = (lane >> 4) * 8;

    const _Float16* Arow[MT];
#pragma unroll
    for (int s = 0; s < MT; ++s)
        Arow[s] = A + (size_t)(tm + s * 16 + (lane & 15)) * lda;
    const _Float16* Wrow[NT];
#pragma unroll
    for (int t = 0; t < NT; ++t)
        Wrow[t] = W + (size_t)(tn + t * 16 + (lane & 15)) * ldw;

    v8f zero = {};
    v8f acc[MT][NT];
#pragma unroll
    for (int s = 0; s < MT; ++s)
#pragma unroll
        for (int t = 0; t < NT; ++t) acc[s][t] = zero;

    for (int k0 = 0; k0 < K; k0 += 32) {
        v16h af[MT];
#pragma unroll
        for (int s = 0; s < MT; ++s) {
            const v8h* pa = (const v8h*)(Arow[s] + k0 + kbaseA);
            v8h lo = pa[0];          // K = kb+0..7   (16B load)
            v8h hi = pa[2];          // K = kb+16..23 (16B load)
            af[s] = __builtin_shufflevector(lo, hi, 0, 1, 2, 3, 4, 5, 6, 7,
                                            8, 9, 10, 11, 12, 13, 14, 15);
        }
#pragma unroll
        for (int t = 0; t < NT; ++t) {
            const v8h* pb = (const v8h*)(Wrow[t] + k0 + kbaseB);
            v8h blo = pb[0], bhi = pb[2];          // K = kb2+0..7, kb2+16..23
            v16h bf = __builtin_shufflevector(blo, bhi, 0, 1, 2, 3, 4, 5, 6, 7,
                                              8, 9, 10, 11, 12, 13, 14, 15);
#pragma unroll
            for (int s = 0; s < MT; ++s)
                acc[s][t] = __builtin_amdgcn_wmma_f32_16x16x32_f16(
                                false, af[s], false, bf, (short)0, acc[s][t],
                                false, false);
        }
    }

#pragma unroll
    for (int s = 0; s < MT; ++s) {
        const int mbase = tm + s * 16 + ((lane >> 4) * 8);
#pragma unroll
        for (int t = 0; t < NT; ++t) {
            const int n = tn + t * 16 + (lane & 15);
            const float bv = bias[n];
#pragma unroll
            for (int j = 0; j < 8; ++j) {
                float v = acc[s][t][j] + bv;
                if (ACT == 1)      v = 0.5f * v * (1.0f + erff(v * 0.70710678118654752f));
                else if (ACT == 2) v = fmaxf(v, 0.0f);
                else if (ACT == 3) v = 1.0f / (1.0f + expf(-v));
                size_t oi = (size_t)(mbase + j) * ldo + n;
                if (OMODE & 1) OUT32[oi] = v;
                if (OMODE & 2) OUT16[oi] = (_Float16)v;
            }
        }
    }
}

// --------------------------- helpers --------------------------------
__device__ inline void atomicMaxF(float* addr, float val) {
    unsigned int* ua = (unsigned int*)addr;
    unsigned int old = *ua;
    while (true) {
        float cur = __uint_as_float(old);
        if (cur >= val) break;
        unsigned int assumed = old;
        old = atomicCAS(ua, assumed, __float_as_uint(val));
        if (old == assumed) break;
    }
}

__global__ void gast_fill(float* __restrict__ p, float v, long n) {
    long i = (long)blockIdx.x * blockDim.x + threadIdx.x;
    long stride = (long)gridDim.x * blockDim.x;
    for (; i < n; i += stride) p[i] = v;
}

// f32 -> f16 conversion (weights)
__global__ void gast_cvt16(const float* __restrict__ in, _Float16* __restrict__ out, int n) {
    int i = blockIdx.x * blockDim.x + threadIdx.x;
    if (i < n) out[i] = (_Float16)in[i];
}

// W1 (D, CIN) -> f16 (D, CPAD) zero-padded
__global__ void gast_pad_w1(const float* __restrict__ w, _Float16* __restrict__ wp, int total) {
    int idx = blockIdx.x * blockDim.x + threadIdx.x;
    if (idx >= total) return;
    int c = idx % CPAD;
    int r = idx / CPAD;
    wp[idx] = (c < CIN) ? (_Float16)w[r * CIN + c] : (_Float16)0.0f;
}

// x (B,C,K,K) -> f16 xT [(b*N+n), c] zero-padded to CPAD
__global__ void gast_permute_x(const float* __restrict__ x, _Float16* __restrict__ xT, int total) {
    int idx = blockIdx.x * blockDim.x + threadIdx.x;
    if (idx >= total) return;
    int c = idx % CPAD;
    int m = idx / CPAD;
    int b = m / NNODE, n = m % NNODE;
    xT[idx] = (c < CIN) ? (_Float16)x[((size_t)(b * CIN + c)) * NNODE + n] : (_Float16)0.0f;
}

// h0 += pos; also emit f16 mirror
__global__ void gast_add_pos(float* __restrict__ h, const float* __restrict__ pos,
                             _Float16* __restrict__ h16, int total) {
    int idx = blockIdx.x * blockDim.x + threadIdx.x;
    if (idx >= total) return;
    int n = (idx / DDIM) % NNODE;
    float v = h[idx] + pos[n * DDIM + (idx & (DDIM - 1))];
    h[idx] = v;
    h16[idx] = (_Float16)v;
}

// flash-style attention, one block per (b, head); output f16 (feeds GEMM)
__global__ void gast_attn(const float* __restrict__ qkv, _Float16* __restrict__ o) {
    const int bh = blockIdx.x;
    const int b = bh / THEAD, h = bh % THEAD;
    __shared__ float kk[NNODE * 8];
    __shared__ float vv[NNODE * 8];
    for (int i = threadIdx.x; i < NNODE * 8; i += blockDim.x) {
        int n = i >> 3, j = i & 7;
        size_t base = ((size_t)(b * NNODE + n)) * (3 * DDIM);
        kk[i] = qkv[base + DDIM     + h * 8 + j];
        vv[i] = qkv[base + 2 * DDIM + h * 8 + j];
    }
    __syncthreads();
    int n = threadIdx.x;
    if (n < NNODE) {
        float q[8];
        size_t base = ((size_t)(b * NNODE + n)) * (3 * DDIM);
#pragma unroll
        for (int j = 0; j < 8; ++j) q[j] = qkv[base + h * 8 + j];
        float mcur = -INFINITY, l = 0.0f;
        float accv[8] = {0, 0, 0, 0, 0, 0, 0, 0};
        for (int m2 = 0; m2 < NNODE; ++m2) {
            float s = 0.0f;
#pragma unroll
            for (int j = 0; j < 8; ++j) s += q[j] * kk[m2 * 8 + j];
            s *= 0.35355339059327379f;              // 1/sqrt(8)
            float mn = fmaxf(mcur, s);
            float corr = expf(mcur - mn);           // exp(-inf)=0 on first iter
            float p = expf(s - mn);
            l = l * corr + p;
#pragma unroll
            for (int j = 0; j < 8; ++j) accv[j] = accv[j] * corr + p * vv[m2 * 8 + j];
            mcur = mn;
        }
        float inv = 1.0f / l;
#pragma unroll
        for (int j = 0; j < 8; ++j)
            o[((size_t)(b * NNODE + n)) * DDIM + h * 8 + j] = (_Float16)(accv[j] * inv);
    }
}

// LayerNorm over rows of 64, one wave per row; optional residual; f32 + optional f16 out
__global__ void gast_ln(const float* __restrict__ x, const float* __restrict__ res,
                        const float* __restrict__ g, const float* __restrict__ bb,
                        float* __restrict__ out, _Float16* __restrict__ out16, int M) {
    int row = blockIdx.x * (blockDim.x >> 5) + (threadIdx.x >> 5);
    int lane = threadIdx.x & 31;
    if (row >= M) return;
    size_t base = (size_t)row * DDIM;
    float e0 = x[base + lane], e1 = x[base + lane + 32];
    if (res) { e0 += res[base + lane]; e1 += res[base + lane + 32]; }
    float s = e0 + e1;
#pragma unroll
    for (int o = 16; o > 0; o >>= 1) s += __shfl_xor(s, o, 32);
    float mean = s * (1.0f / 64.0f);
    float d0 = e0 - mean, d1 = e1 - mean;
    float v = d0 * d0 + d1 * d1;
#pragma unroll
    for (int o = 16; o > 0; o >>= 1) v += __shfl_xor(v, o, 32);
    float rstd = rsqrtf(v * (1.0f / 64.0f) + 1e-5f);
    float r0 = d0 * rstd * g[lane]      + bb[lane];
    float r1 = d1 * rstd * g[lane + 32] + bb[lane + 32];
    out[base + lane]      = r0;
    out[base + lane + 32] = r1;
    if (out16) {
        out16[base + lane]      = (_Float16)r0;
        out16[base + lane + 32] = (_Float16)r1;
    }
}

// GAT: per-(b,e,h) logit + scatter-max over dst (xl/xr stored f16)
__global__ void gast_gat_logit(const _Float16* __restrict__ xl, const _Float16* __restrict__ xr,
                               const float* __restrict__ att,
                               const int* __restrict__ src, const int* __restrict__ dst,
                               float* __restrict__ logit, float* __restrict__ mbuf, int E) {
    int idx = blockIdx.x * blockDim.x + threadIdx.x;
    int total = BATCH * E * GHEAD;
    if (idx >= total) return;
    int h = idx & 7;
    int t = idx >> 3;
    int e = t % E, b = t / E;
    int sN = src[e], dN = dst[e];
    const _Float16* pl = xl + ((size_t)(b * NNODE + sN)) * (GHEAD * DDIM) + h * DDIM;
    const _Float16* pr = xr + ((size_t)(b * NNODE + dN)) * (GHEAD * DDIM) + h * DDIM;
    const float* pa = att + h * DDIM;
    float acc = 0.0f;
    for (int d = 0; d < DDIM; ++d) {
        float tt = (float)pl[d] + (float)pr[d];
        tt = (tt > 0.0f) ? tt : 0.2f * tt;      // leaky_relu 0.2
        acc += tt * pa[d];
    }
    logit[idx] = acc;
    atomicMaxF(mbuf + ((size_t)(b * NNODE + dN)) * GHEAD + h, acc);
}

__global__ void gast_gat_alpha(float* __restrict__ logit, const float* __restrict__ mbuf,
                               float* __restrict__ denom, const int* __restrict__ dst, int E) {
    int idx = blockIdx.x * blockDim.x + threadIdx.x;
    int total = BATCH * E * GHEAD;
    if (idx >= total) return;
    int h = idx & 7;
    int t = idx >> 3;
    int e = t % E, b = t / E;
    int dN = dst[e];
    size_t mi = ((size_t)(b * NNODE + dN)) * GHEAD + h;
    float a = expf(logit[idx] - mbuf[mi]);
    logit[idx] = a;                              // alpha in place
    atomicAdd(denom + mi, a);
}

// out[b,dst,:] += (alpha/denom)/GH * xl[b,src,h,:]
__global__ void gast_gat_scatter(const float* __restrict__ alpha, const float* __restrict__ denom,
                                 const _Float16* __restrict__ xl,
                                 const int* __restrict__ src, const int* __restrict__ dst,
                                 float* __restrict__ out, int E) {
    int idx = blockIdx.x * blockDim.x + threadIdx.x;
    int total = BATCH * E * GHEAD;
    if (idx >= total) return;
    int h = idx & 7;
    int t = idx >> 3;
    int e = t % E, b = t / E;
    int sN = src[e], dN = dst[e];
    size_t mi = ((size_t)(b * NNODE + dN)) * GHEAD + h;
    float w = alpha[idx] / denom[mi] * (1.0f / (float)GHEAD);
    const _Float16* pl = xl + ((size_t)(b * NNODE + sN)) * (GHEAD * DDIM) + h * DDIM;
    float* po = out + ((size_t)(b * NNODE + dN)) * DDIM;
    for (int d = 0; d < DDIM; ++d) atomicAdd(po + d, w * (float)pl[d]);
}

// relu(acc + bias) in place + f16 mirror (feeds next layer's GEMM)
__global__ void gast_gat_finish(float* __restrict__ acc, const float* __restrict__ bias,
                                _Float16* __restrict__ out16, int total) {
    int idx = blockIdx.x * blockDim.x + threadIdx.x;
    if (idx >= total) return;
    float v = fmaxf(acc[idx] + bias[idx & (DDIM - 1)], 0.0f);
    acc[idx] = v;
    out16[idx] = (_Float16)v;
}

// gate path only at node n = N//2 = 60; concat row in f16 (feeds GEMM)
__global__ void gast_cat_row(const float* __restrict__ spec, const float* __restrict__ spat,
                             _Float16* __restrict__ cat) {
    int idx = blockIdx.x * blockDim.x + threadIdx.x;
    if (idx >= BATCH * 2 * DDIM) return;
    int b = idx >> 7, d = idx & 127;
    size_t base = ((size_t)(b * NNODE + NNODE / 2)) * DDIM;
    cat[idx] = (_Float16)((d < DDIM) ? spec[base + d] : spat[base + d - DDIM]);
}

__global__ void gast_fuse_row(const float* __restrict__ gate, const float* __restrict__ spec,
                              const float* __restrict__ spat, float* __restrict__ fused) {
    int idx = blockIdx.x * blockDim.x + threadIdx.x;
    if (idx >= BATCH * DDIM) return;
    int b = idx >> 6, d = idx & 63;
    size_t base = ((size_t)(b * NNODE + NNODE / 2)) * DDIM;
    float g = gate[idx];
    fused[idx] = g * spat[base + d] + (1.0f - g) * spec[base + d];
}

// ---------------------------------------------------------------------
extern "C" void kernel_launch(void* const* d_in, const int* in_sizes, int n_in,
                              void* d_out, int out_size, void* d_ws, size_t ws_size,
                              hipStream_t stream) {
    (void)n_in; (void)out_size; (void)ws_size;
    // ---- inputs in setup_inputs() dict order ----
    const float* x       = (const float*)d_in[0];
    const int*   eidx    = (const int*)  d_in[1];
    const int    E       = in_sizes[1] / 2;
    const int* src = eidx;
    const int* dst = eidx + E;
    const float* sW1  = (const float*)d_in[2];
    const float* sb1  = (const float*)d_in[3];
    const float* sW2  = (const float*)d_in[4];
    const float* sb2  = (const float*)d_in[5];
    const float* spos = (const float*)d_in[6];
    const float* Wqkv = (const float*)d_in[7];
    const float* bqkv = (const float*)d_in[8];
    const float* Wo   = (const float*)d_in[9];
    const float* bo   = (const float*)d_in[10];
    const float* g1   = (const float*)d_in[11];
    const float* b1n  = (const float*)d_in[12];
    const float* Wf1  = (const float*)d_in[13];
    const float* bf1  = (const float*)d_in[14];
    const float* Wf2  = (const float*)d_in[15];
    const float* bf2  = (const float*)d_in[16];
    const float* g2   = (const float*)d_in[17];
    const float* b2n  = (const float*)d_in[18];
    const float* gateW = (const float*)d_in[43];
    const float* gateB = (const float*)d_in[44];
    const float* clsG  = (const float*)d_in[45];
    const float* clsB  = (const float*)d_in[46];
    const float* clsW1 = (const float*)d_in[47];
    const float* clsb1 = (const float*)d_in[48];
    const float* clsW2 = (const float*)d_in[49];
    const float* clsb2 = (const float*)d_in[50];
    float* out = (float*)d_out;

    // ---- workspace layout (unit = float, all chunks 16B aligned) ----
    const size_t M = MROWS;
    float* ws = (float*)d_ws;
    size_t off = 0;
    auto afl = [&](size_t nfl) { float* p = ws + off; off += (nfl + 7) & ~(size_t)7; return p; };
    auto ah  = [&](size_t nh)  { return (_Float16*)afl((nh + 1) / 2); };

    // f32 buffers
    float* h0    = afl(M * 64);
    float* qkv   = afl(M * 192);
    float* aproj = afl(M * 64);
    float* h1    = afl(M * 64);
    float* f2    = afl(M * 64);
    float* spec  = afl(M * 64);
    float* hA    = afl(M * 64);
    float* hB    = afl(M * 64);
    float* logit = afl((size_t)BATCH * E * GHEAD);
    float* mbuf  = afl((size_t)BATCH * NNODE * GHEAD);
    float* denom = afl((size_t)BATCH * NNODE * GHEAD);
    float* gateR = afl(BATCH * 64);
    float* fusedR= afl(BATCH * 64);
    float* lnr32 = afl(BATCH * 64);
    // f16 buffers
    _Float16* xTh   = ah((size_t)M * CPAD);
    _Float16* t1h   = ah(M * 64);
    _Float16* h0h   = ah(M * 64);
    _Float16* obufh = ah(M * 64);
    _Float16* h1h   = ah(M * 64);
    _Float16* f1h   = ah(M * 128);
    _Float16* spech = ah(M * 64);
    _Float16* xlh   = ah(M * 512);
    _Float16* xrh   = ah(M * 512);
    _Float16* hAh   = ah(M * 64);
    _Float16* hBh   = ah(M * 64);
    _Float16* w1ph  = ah(DDIM * CPAD);
    _Float16* w2h   = ah(64 * 64);
    _Float16* wqkvh = ah(192 * 64);
    _Float16* woh   = ah(64 * 64);
    _Float16* wf1h  = ah(128 * 64);
    _Float16* wf2h  = ah(64 * 128);
    _Float16* wlh   = ah(512 * 64);
    _Float16* wrh   = ah(512 * 64);
    _Float16* gatewh= ah(64 * 128);
    _Float16* clsw1h= ah(128 * 64);
    _Float16* clsw2h= ah(16 * 128);
    _Float16* cath  = ah(BATCH * 128);
    _Float16* lnrh  = ah(BATCH * 64);
    _Float16* c1h   = ah(BATCH * 128);

    const int TPB = 256;
    const int MT2  = (int)(M / 32);     // 968 row-tiles (MT=2)
    const int nEdgeTh = BATCH * E * GHEAD;
    auto cvt = [&](const float* s, _Float16* d2, int n) {
        gast_cvt16<<<(n + TPB - 1) / TPB, TPB, 0, stream>>>(s, d2, n);
    };

    // ---- weight conversions (f32 -> f16) ----
    {
        int totw = DDIM * CPAD;
        gast_pad_w1<<<(totw + TPB - 1) / TPB, TPB, 0, stream>>>(sW1, w1ph, totw);
    }
    cvt(sW2, w2h, 64 * 64);       cvt(Wqkv, wqkvh, 192 * 64);
    cvt(Wo, woh, 64 * 64);        cvt(Wf1, wf1h, 128 * 64);
    cvt(Wf2, wf2h, 64 * 128);     cvt(gateW, gatewh, 64 * 128);
    cvt(clsW1, clsw1h, 128 * 64); cvt(clsW2, clsw2h, 16 * 128);

    // ---- spectral MLP ----
    {
        int tot = (int)(M * CPAD);
        gast_permute_x<<<(tot + TPB - 1) / TPB, TPB, 0, stream>>>(x, xTh, tot);
    }
    gast_gemm_wmma<4, 2, 1, 2><<<dim3(1, MT2), 32, 0, stream>>>(xTh, w1ph, sb1, nullptr, t1h,
        CPAD, CPAD, CPAD, DDIM);
    gast_gemm_wmma<4, 2, 0, 1><<<dim3(1, MT2), 32, 0, stream>>>(t1h, w2h, sb2, h0, nullptr,
        DDIM, DDIM, DDIM, DDIM);
    {
        int tot = (int)(M * DDIM);
        gast_add_pos<<<(tot + TPB - 1) / TPB, TPB, 0, stream>>>(h0, spos, h0h, tot);
    }

    // ---- transformer block ----
    gast_gemm_wmma<4, 2, 0, 1><<<dim3(3, MT2), 32, 0, stream>>>(h0h, wqkvh, bqkv, qkv, nullptr,
        DDIM, DDIM, DDIM, 3 * DDIM);
    gast_attn<<<BATCH * THEAD, 128, 0, stream>>>(qkv, obufh);
    gast_gemm_wmma<4, 2, 0, 1><<<dim3(1, MT2), 32, 0, stream>>>(obufh, woh, bo, aproj, nullptr,
        DDIM, DDIM, DDIM, DDIM);
    gast_ln<<<(int)(M / 8), 256, 0, stream>>>(h0, aproj, g1, b1n, h1, h1h, (int)M);
    gast_gemm_wmma<4, 2, 2, 2><<<dim3(2, MT2), 32, 0, stream>>>(h1h, wf1h, bf1, nullptr, f1h,
        DDIM, DDIM, DDIM, FFDIM);
    gast_gemm_wmma<4, 2, 0, 1><<<dim3(1, MT2), 32, 0, stream>>>(f1h, wf2h, bf2, f2, nullptr,
        FFDIM, FFDIM, FFDIM, DDIM);
    gast_ln<<<(int)(M / 8), 256, 0, stream>>>(h1, f2, g2, b2n, spec, spech, (int)M);

    // ---- GAT layers ----
    const _Float16* hinH = spech;
    float*    houts[NLAY]  = {hA, hB, hA, hB};
    _Float16* houtsH[NLAY] = {hAh, hBh, hAh, hBh};
    for (int l = 0; l < NLAY; ++l) {
        const float* Wl   = (const float*)d_in[19 + 6 * l + 0];
        const float* bl   = (const float*)d_in[19 + 6 * l + 1];
        const float* Wr   = (const float*)d_in[19 + 6 * l + 2];
        const float* br   = (const float*)d_in[19 + 6 * l + 3];
        const float* attW = (const float*)d_in[19 + 6 * l + 4];
        const float* bias = (const float*)d_in[19 + 6 * l + 5];

        cvt(Wl, wlh, 512 * 64);
        cvt(Wr, wrh, 512 * 64);
        gast_gemm_wmma<4, 2, 0, 2><<<dim3(8, MT2), 32, 0, stream>>>(hinH, wlh, bl, nullptr, xlh,
            DDIM, DDIM, DDIM, GHEAD * DDIM);
        gast_gemm_wmma<4, 2, 0, 2><<<dim3(8, MT2), 32, 0, stream>>>(hinH, wrh, br, nullptr, xrh,
            DDIM, DDIM, DDIM, GHEAD * DDIM);

        long nM = (long)BATCH * NNODE * GHEAD;
        gast_fill<<<1024, TPB, 0, stream>>>(mbuf, -INFINITY, nM);
        gast_fill<<<1024, TPB, 0, stream>>>(denom, 0.0f, nM);
        gast_fill<<<2048, TPB, 0, stream>>>(houts[l], 0.0f, (long)M * DDIM);

        gast_gat_logit<<<(nEdgeTh + TPB - 1) / TPB, TPB, 0, stream>>>(xlh, xrh, attW,
            src, dst, logit, mbuf, E);
        gast_gat_alpha<<<(nEdgeTh + TPB - 1) / TPB, TPB, 0, stream>>>(logit, mbuf, denom, dst, E);
        gast_gat_scatter<<<(nEdgeTh + TPB - 1) / TPB, TPB, 0, stream>>>(logit, denom, xlh,
            src, dst, houts[l], E);
        {
            int tot = (int)(M * DDIM);
            gast_gat_finish<<<(tot + TPB - 1) / TPB, TPB, 0, stream>>>(houts[l], bias,
                houtsH[l], tot);
        }
        hinH = houtsH[l];
    }
    const float* spat = houts[NLAY - 1];   // hB (f32)

    // ---- gate + fuse (only node N/2) ----
    gast_cat_row<<<(BATCH * 128 + TPB - 1) / TPB, TPB, 0, stream>>>(spec, spat, cath);
    gast_gemm_wmma<4, 2, 3, 1><<<dim3(1, BATCH / 32), 32, 0, stream>>>(cath, gatewh, gateB,
        gateR, nullptr, 2 * DDIM, 2 * DDIM, 2 * DDIM, DDIM);
    gast_fuse_row<<<(BATCH * DDIM + TPB - 1) / TPB, TPB, 0, stream>>>(gateR, spec, spat, fusedR);

    // ---- classifier ----
    gast_ln<<<BATCH / 8, 256, 0, stream>>>(fusedR, (const float*)nullptr, clsG, clsB,
        lnr32, lnrh, BATCH);
    gast_gemm_wmma<4, 2, 1, 2><<<dim3(2, BATCH / 32), 32, 0, stream>>>(lnrh, clsw1h, clsb1,
        nullptr, c1h, DDIM, DDIM, DDIM, 2 * DDIM);
    gast_gemm_wmma<1, 2, 0, 1><<<dim3(1, BATCH / 32), 32, 0, stream>>>(c1h, clsw2h, clsb2,
        out, nullptr, 2 * DDIM, 2 * DDIM, 2 * DDIM, NCLS);
}
